// Model_8813272891895
// MI455X (gfx1250) — matrix operand order repaired and checked by reading, not compiled, not run on hardware
//
#include <hip/hip_runtime.h>
#include <math.h>

typedef __attribute__((ext_vector_type(16))) _Float16 v16h;
typedef __attribute__((ext_vector_type(8)))  float    v8f;

struct alignas(16) U4 { unsigned int a, b, c, d; };
struct alignas(8)  U2 { unsigned int a, b; };

union AFrag {
    v16h     v;
    _Float16 h[16];
    U4       q[2];
};

// ---------------------------------------------------------------------------
// A-fragment gather: 8 contiguous K-halves for one output position.
// Patch K ordering: (kh, kw, c) with c over CP (padded channels), NHWC input.
// base = in + n*Hin*Win*CP, ihb/iwb = top-left of receptive field.
// With the K loop fully unrolled, cell/kh/kw/c0 are compile-time constants.
// ---------------------------------------------------------------------------
template<int KH, int KW, int CP, bool PRER>
__device__ __forceinline__ void load_a8(const _Float16* __restrict__ base,
                                        int ihb, int iwb, int k,
                                        int Hin, int Win,
                                        _Float16* __restrict__ d)
{
    if constexpr (CP >= 16) {
        const int cell = k / CP, c0 = k % CP;
        const int ih = ihb + cell / KW;
        const int iw = iwb + cell % KW;
        U4 q = {0u, 0u, 0u, 0u};
        if ((unsigned)ih < (unsigned)Hin && (unsigned)iw < (unsigned)Win)
            q = *(const U4*)&base[((size_t)ih * Win + iw) * CP + c0];
        *(U4*)d = q;
    } else {
        // CP == 4, KW == 4 (first conv): one 8-run spans two (kh,kw) cells.
        const int cell = k >> 2;
        const int ih  = ihb + (cell >> 2);
        const int kw0 = cell & 3;
#pragma unroll
        for (int j = 0; j < 2; ++j) {
            const int iw = iwb + kw0 + j;
            U2 q = {0u, 0u};
            if ((unsigned)ih < (unsigned)Hin && (unsigned)iw < (unsigned)Win)
                q = *(const U2*)&base[((size_t)ih * Win + iw) * CP];
            *(U2*)(d + 4 * j) = q;
        }
    }
    if constexpr (PRER) {
#pragma unroll
        for (int i = 0; i < 8; ++i)
            d[i] = (d[i] > (_Float16)0.f) ? d[i] : (_Float16)0.f;
    }
}

// ---------------------------------------------------------------------------
// Implicit-GEMM convolution via v_wmma_f32_16x16x32_f16.
// in  : NHWC f16 (CP channels), wgt: [COUT][K] f16, out: NHWC f16 (COUT).
// ---------------------------------------------------------------------------
template<int KH, int KW, int CP, int COUT, int BN, int BM, int STRIDE, int PAD,
         bool PRER, bool RELU_, bool RES>
__global__ __launch_bounds__(256)
void conv_wmma(const _Float16* __restrict__ in, const _Float16* __restrict__ wgt,
               const float* __restrict__ bias, const _Float16* __restrict__ resid,
               _Float16* __restrict__ out,
               int N, int Hin, int Win, int Hout, int Wout)
{
    constexpr int K  = KH * KW * CP;
    constexpr int NT = BN / 16;
    constexpr int MT = BM / 16;
    constexpr int MS = (MT * NT) / 8;
    constexpr int WSTEP = 8 / NT;
    (void)N;

    const int tid  = threadIdx.x;
    const int lane = tid & 31;
    const int wave = tid >> 5;
    const int wn   = wave % NT;
    const int wm0  = wave / NT;
    const int mblk = blockIdx.x * BM;
    const int col  = blockIdx.y * BN + wn * 16 + (lane & 15);

    // Per-subtile row decode hoisted out of the K loop.
    int ihb[MS], iwb[MS];
    const _Float16* abase[MS];
#pragma unroll
    for (int s = 0; s < MS; ++s) {
        const int sm = wm0 + s * WSTEP;
        const int m  = mblk + sm * 16 + (lane & 15);
        const int ow = m % Wout;
        const int t1 = m / Wout;
        const int oh = t1 % Hout;
        const int n  = t1 / Hout;
        ihb[s] = oh * STRIDE - PAD;
        iwb[s] = ow * STRIDE - PAD;
        abase[s] = in + (size_t)n * Hin * Win * CP;
    }
    const _Float16* wbase = wgt + (size_t)col * K;

    v8f zero = {};
    v8f acc[MS];
#pragma unroll
    for (int s = 0; s < MS; ++s) acc[s] = zero;

#pragma unroll
    for (int k0 = 0; k0 < K; k0 += 32) {
        AFrag bf;
        {
            const int kb = k0 + ((lane >> 4) << 3);
            bf.q[0] = *(const U4*)(wbase + kb);
            bf.q[1] = *(const U4*)(wbase + kb + 16);
        }
#pragma unroll
        for (int s = 0; s < MS; ++s) {
            AFrag af;
            const int kb = k0 + ((lane >> 4) << 3);
            load_a8<KH, KW, CP, PRER>(abase[s], ihb[s], iwb[s], kb,      Hin, Win, &af.h[0]);
            load_a8<KH, KW, CP, PRER>(abase[s], ihb[s], iwb[s], kb + 16, Hin, Win, &af.h[8]);
            acc[s] = __builtin_amdgcn_wmma_f32_16x16x32_f16(false, af.v, false, bf.v,
                                                            (short)0, acc[s], false, false);
        }
    }

    const float b = bias ? bias[col] : 0.f;
#pragma unroll
    for (int s = 0; s < MS; ++s) {
        const int sm = wm0 + s * WSTEP;
        const size_t idx0 = (size_t)(mblk + sm * 16 + ((lane >> 4) << 3)) * COUT + col;
#pragma unroll
        for (int r = 0; r < 8; ++r) {
            float v = acc[s][r] + b;
            if constexpr (RES)   v += (float)resid[idx0 + (size_t)r * COUT];
            if constexpr (RELU_) v = v > 0.f ? v : 0.f;
            out[idx0 + (size_t)r * COUT] = (_Float16)v;
        }
    }
}

// ---------------------------------------------------------------------------
// Transposed conv (k=4, s=2, p=1) via parity decomposition: each output uses
// exactly 2x2 taps; weights pre-packed per parity as [4][COUTP][4*CP].
// Block covers one output row (uniform oh) and one ow parity (blockIdx.y).
// ---------------------------------------------------------------------------
template<int CP, int COUT, int COUTP, int BM, bool PRER, bool RELU_, bool F32OUT>
__global__ __launch_bounds__(256)
void convt_wmma(const _Float16* __restrict__ in, const _Float16* __restrict__ wgt,
                const float* __restrict__ bias,
                _Float16* __restrict__ out, float* __restrict__ out32,
                int N, int Hin, int Win, int Hout, int Wout)
{
    constexpr int K  = 4 * CP;
    constexpr int BN = (COUTP < 64) ? COUTP : 64;
    constexpr int NT = BN / 16;
    constexpr int MT = BM / 16;
    constexpr int MS = (MT * NT) / 8;
    constexpr int WSTEP = 8 / NT;
    (void)N;

    const int tid = threadIdx.x, lane = tid & 31, wave = tid >> 5;
    const int wn = wave % NT, wm0 = wave / NT;
    const int px   = blockIdx.y;
    const int Wh   = Wout >> 1;
    const int mblk = blockIdx.x * BM;
    const int oh_b = (mblk / Wh) % Hout;      // uniform across block (BM <= Wh)
    const int n_b  = mblk / (Wh * Hout);
    const int ph   = (oh_b + 1) & 1;
    const int pw   = (px + 1) & 1;
    const int ih0  = (oh_b + 1 - ph) >> 1;
    const _Float16* wsel = wgt + (size_t)(ph * 2 + pw) * COUTP * K;
    const int col = blockIdx.z * BN + wn * 16 + (lane & 15);
    const _Float16* abase = in + (size_t)n_b * Hin * Win * CP;
    const _Float16* wbase = wsel + (size_t)col * K;

    // Per-subtile column decode hoisted out of the K loop.
    int iwb[MS];
#pragma unroll
    for (int s = 0; s < MS; ++s) {
        const int sm  = wm0 + s * WSTEP;
        const int m   = mblk + sm * 16 + (lane & 15);
        const int owp = m % Wh;
        const int ow  = owp * 2 + px;
        iwb[s] = (ow + 1 - pw) >> 1;
    }

    v8f zero = {};
    v8f acc[MS];
#pragma unroll
    for (int s = 0; s < MS; ++s) acc[s] = zero;

#pragma unroll
    for (int k0 = 0; k0 < K; k0 += 32) {
        AFrag bf;
        {
            const int kb = k0 + ((lane >> 4) << 3);
            bf.q[0] = *(const U4*)(wbase + kb);
            bf.q[1] = *(const U4*)(wbase + kb + 16);
        }
#pragma unroll
        for (int s = 0; s < MS; ++s) {
            AFrag af;
#pragma unroll
            for (int half = 0; half < 2; ++half) {
                const int k  = k0 + ((lane >> 4) << 3) + half * 16;
                const int t  = k / CP, c0 = k % CP;   // constants after unroll
                const int th = t >> 1, tw = t & 1;
                const int ih = ih0 - th;
                const int iw = iwb[s] - tw;
                U4 q = {0u, 0u, 0u, 0u};
                if ((unsigned)ih < (unsigned)Hin && (unsigned)iw < (unsigned)Win)
                    q = *(const U4*)&abase[((size_t)ih * Win + iw) * CP + c0];
                *(U4*)&af.h[half * 8] = q;
            }
            if constexpr (PRER) {
#pragma unroll
                for (int i = 0; i < 16; ++i)
                    af.h[i] = (af.h[i] > (_Float16)0.f) ? af.h[i] : (_Float16)0.f;
            }
            acc[s] = __builtin_amdgcn_wmma_f32_16x16x32_f16(false, af.v, false, bf.v,
                                                            (short)0, acc[s], false, false);
        }
    }

    const float b = (col < COUT && bias) ? bias[col] : 0.f;
#pragma unroll
    for (int s = 0; s < MS; ++s) {
        const int sm = wm0 + s * WSTEP;
        const int m0 = mblk + sm * 16 + ((lane >> 4) << 3);
        const int owp0 = m0 % Wh;
#pragma unroll
        for (int r = 0; r < 8; ++r) {
            const int ow = (owp0 + r) * 2 + px;
            float v = acc[s][r] + b;
            if constexpr (RELU_) v = v > 0.f ? v : 0.f;
            if constexpr (F32OUT) {
                if (col < COUT)
                    out32[(((size_t)n_b * COUT + col) * Hout + oh_b) * Wout + ow] = v;
            } else {
                out[(((size_t)n_b * Hout + oh_b) * Wout + ow) * COUT + col] = (_Float16)v;
            }
        }
    }
}

// ---------------------------------------------------------------------------
// VQ: per block, 16 flat z-rows vs all 512 codes. dot via WMMA (K=64),
// argmin over (|c|^2 - 2 z.c) with ds_min_u64 key = (orderable_f32<<32)|idx.
// ---------------------------------------------------------------------------
__global__ __launch_bounds__(256)
void vq_kernel(const _Float16* __restrict__ z, const _Float16* __restrict__ cb,
               const float* __restrict__ cnorm,
               _Float16* __restrict__ q, float* __restrict__ idx_out,
               int* __restrict__ counts, float* __restrict__ sse)
{
    __shared__ unsigned long long best[16];
    __shared__ int   sidx[16];
    __shared__ float ssum;

    const int tid = threadIdx.x, lane = tid & 31, wave = tid >> 5;
    if (tid < 16) best[tid] = ~0ULL;
    if (tid == 0) ssum = 0.f;
    __syncthreads();

    const int rowb = blockIdx.x * 16;

    v8f zero = {};
    v8f acc[4];
#pragma unroll
    for (int s = 0; s < 4; ++s) acc[s] = zero;

#pragma unroll
    for (int k0 = 0; k0 < 64; k0 += 32) {
        AFrag af;
        {
            const int row = rowb + (lane & 15);
            const int kb  = k0 + ((lane >> 4) << 3);
            const _Float16* zp = z + (size_t)row * 64 + kb;
            af.q[0] = *(const U4*)zp;
            af.q[1] = *(const U4*)(zp + 16);
        }
#pragma unroll
        for (int s = 0; s < 4; ++s) {
            const int c  = wave * 64 + s * 16 + (lane & 15);
            const int kb = k0 + ((lane >> 4) << 3);
            AFrag bf;
            const _Float16* cp = cb + (size_t)c * 64 + kb;
            bf.q[0] = *(const U4*)cp;
            bf.q[1] = *(const U4*)(cp + 16);
            acc[s] = __builtin_amdgcn_wmma_f32_16x16x32_f16(false, af.v, false, bf.v,
                                                            (short)0, acc[s], false, false);
        }
    }

#pragma unroll
    for (int s = 0; s < 4; ++s) {
        const int c  = wave * 64 + s * 16 + (lane & 15);
        const float cn = cnorm[c];
#pragma unroll
        for (int r = 0; r < 8; ++r) {
            const int rl = ((lane >> 4) << 3) + r;
            const float score = cn - 2.f * acc[s][r];
            unsigned u = __float_as_uint(score);
            u = (u & 0x80000000u) ? ~u : (u | 0x80000000u);
            const unsigned long long key = ((unsigned long long)u << 32) | (unsigned)c;
            atomicMin(&best[rl], key);
        }
    }
    __syncthreads();

    if (tid < 16) {
        const int c = (int)(best[tid] & 0xFFFFFFFFu);
        sidx[tid] = c;
        idx_out[rowb + tid] = (float)c;
        atomicAdd(&counts[c], 1);
    }
    __syncthreads();

    {
        const int rl = tid >> 4, j = tid & 15;
        const int c = sidx[rl];
        const size_t zo = (size_t)(rowb + rl) * 64 + j * 4;
        float part = 0.f;
#pragma unroll
        for (int e = 0; e < 4; ++e) {
            const float qv = (float)cb[(size_t)c * 64 + j * 4 + e];
            const float zv = (float)z[zo + e];
            q[zo + e] = (_Float16)qv;
            const float d = qv - zv;
            part += d * d;
        }
        atomicAdd(&ssum, part);
    }
    __syncthreads();
    if (tid == 0) atomicAdd(sse, ssum);
}

__global__ void vq_finalize(const int* __restrict__ counts, const float* __restrict__ sse,
                            float* __restrict__ loss_out, float* __restrict__ perp_out,
                            float invN, float invElems)
{
    __shared__ float s[512];
    const int t = threadIdx.x;
    const float p = (float)counts[t] * invN;
    s[t] = -p * logf(p + 1e-10f);
    __syncthreads();
    for (int st = 256; st > 0; st >>= 1) {
        if (t < st) s[t] += s[t + st];
        __syncthreads();
    }
    if (t == 0) {
        *perp_out = expf(s[0]);
        *loss_out = 0.25f * (*sse) * invElems;
    }
}

// ---------------------------------------------------------------------------
// Prep kernels
// ---------------------------------------------------------------------------
__global__ void prep_conv_w(const float* __restrict__ w, _Float16* __restrict__ out,
                            int O, int I, int KH, int KW, int CP)
{
    const int i = blockIdx.x * blockDim.x + threadIdx.x;
    const int total = O * KH * KW * CP;
    if (i >= total) return;
    const int cp = i % CP;
    int t = i / CP;
    const int kw = t % KW; t /= KW;
    const int kh = t % KH;
    const int o  = t / KH;
    float v = 0.f;
    if (cp < I) v = w[(((size_t)o * I + cp) * KH + kh) * KW + kw];
    out[i] = (_Float16)v;
}

// torch layout w[I][O][4][4] -> out[ph*2+pw][OP][tap(4)][I], tap=(th,tw), kh=ph+2*th
__global__ void prep_convt_w(const float* __restrict__ w, _Float16* __restrict__ out,
                             int I, int O, int OP)
{
    const int i = blockIdx.x * blockDim.x + threadIdx.x;
    const int total = 4 * OP * 4 * I;
    if (i >= total) return;
    const int c = i % I;
    int t = i / I;
    const int tap = t % 4; t /= 4;
    const int o   = t % OP;
    const int par = t / OP;
    const int pw = par & 1, ph = par >> 1;
    const int th = tap >> 1, tw = tap & 1;
    float v = 0.f;
    if (o < O) v = w[(((size_t)c * O + o) * 4 + (ph + 2 * th)) * 4 + (pw + 2 * tw)];
    out[i] = (_Float16)v;
}

__global__ void prep_codebook(const float* __restrict__ cbf, _Float16* __restrict__ cbh,
                              float* __restrict__ cnorm)
{
    const int c = blockIdx.x * blockDim.x + threadIdx.x;
    if (c >= 512) return;
    float s = 0.f;
    for (int k = 0; k < 64; ++k) {
        const float v = cbf[c * 64 + k];
        s += v * v;
        cbh[c * 64 + k] = (_Float16)v;
    }
    cnorm[c] = s;
}

__global__ void cvt_input(const float* __restrict__ x, _Float16* __restrict__ out,
                          int N, int H, int W)
{
    const size_t i = (size_t)blockIdx.x * blockDim.x + threadIdx.x;
    const size_t total = (size_t)N * H * W;
    if (i >= total) return;
    const int w = (int)(i % W);
    const size_t t = i / W;
    const int h = (int)(t % H);
    const int n = (int)(t / H);
    for (int c = 0; c < 4; ++c) {
        const float v = (c < 3) ? x[(((size_t)n * 3 + c) * H + h) * W + w] : 0.f;
        out[i * 4 + c] = (_Float16)v;
    }
}

__global__ void zero_u32(unsigned int* __restrict__ p, int n)
{
    const int i = blockIdx.x * blockDim.x + threadIdx.x;
    if (i < n) p[i] = 0u;
}

// ---------------------------------------------------------------------------
// Host launcher
// ---------------------------------------------------------------------------
extern "C" void kernel_launch(void* const* d_in, const int* in_sizes, int n_in,
                              void* d_out, int out_size, void* d_ws, size_t ws_size,
                              hipStream_t stream)
{
    (void)in_sizes; (void)n_in; (void)out_size; (void)ws_size;

    const float* x     = (const float*)d_in[0];
    const float* we1   = (const float*)d_in[1];
    const float* be1   = (const float*)d_in[2];
    const float* we2   = (const float*)d_in[3];
    const float* be2   = (const float*)d_in[4];
    const float* we3   = (const float*)d_in[5];
    const float* be3   = (const float*)d_in[6];
    const float* er0w3 = (const float*)d_in[7];
    const float* er0w1 = (const float*)d_in[8];
    const float* er1w3 = (const float*)d_in[9];
    const float* er1w1 = (const float*)d_in[10];
    const float* wpq   = (const float*)d_in[11];
    const float* bpq   = (const float*)d_in[12];
    const float* cbf   = (const float*)d_in[13];
    const float* wd1   = (const float*)d_in[14];
    const float* bd1   = (const float*)d_in[15];
    const float* dr0w3 = (const float*)d_in[16];
    const float* dr0w1 = (const float*)d_in[17];
    const float* dr1w3 = (const float*)d_in[18];
    const float* dr1w1 = (const float*)d_in[19];
    const float* wt1   = (const float*)d_in[20];
    const float* bt1   = (const float*)d_in[21];
    const float* wt2   = (const float*)d_in[22];
    const float* bt2   = (const float*)d_in[23];

    float* out   = (float*)d_out;
    float* recon = out + 1;
    float* perp  = out + 1 + 6291456;
    float* idxf  = out + 1 + 6291456 + 1;

    char* cur = (char*)d_ws;
    auto alloc = [&](size_t bytes) -> void* {
        void* r = (void*)cur;
        cur += (bytes + 255) & ~(size_t)255;
        return r;
    };
    auto a16 = [&](size_t elems) -> _Float16* { return (_Float16*)alloc(elems * 2); };

    // Activations (f16, NHWC)
    _Float16* Xp = a16(8388608);   // 32x256x256x4
    _Float16* A1 = a16(33554432);  // 32x128x128x64
    _Float16* A2 = a16(16777216);  // 32x64x64x128
    _Float16* E  = a16(16777216);  // enc trunk 32x64x64x128
    _Float16* Tt = a16(4194304);   // res scratch 32x64x64x32
    _Float16* Z  = a16(8388608);   // 32x64x64x64
    _Float16* Q  = a16(8388608);   // 32x64x64x64
    _Float16* D0 = a16(16777216);  // dec trunk 32x64x64x128
    _Float16* T1 = a16(33554432);  // 32x128x128x64

    // Weights (f16)
    _Float16* W1   = a16(64 * 64);
    _Float16* W2   = a16(128 * 1024);
    _Float16* W3   = a16(128 * 1152);
    _Float16* ER0A = a16(32 * 1152);
    _Float16* ER0B = a16(128 * 32);
    _Float16* ER1A = a16(32 * 1152);
    _Float16* ER1B = a16(128 * 32);
    _Float16* WPQ  = a16(64 * 128);
    _Float16* CBH  = a16(512 * 64);
    _Float16* WD1  = a16(128 * 576);
    _Float16* DR0A = a16(32 * 1152);
    _Float16* DR0B = a16(128 * 32);
    _Float16* DR1A = a16(32 * 1152);
    _Float16* DR1B = a16(128 * 32);
    _Float16* WT1  = a16(4 * 64 * 512);
    _Float16* WT2  = a16(4 * 16 * 256);

    float* cnorm = (float*)alloc(512 * 4);
    float* stat  = (float*)alloc(516 * 4);   // [0]=sse, [1..512]=counts
    float* sse   = stat;
    int*   counts = (int*)(stat + 1);

    // --- prep ---
    zero_u32<<<dim3(3), 256, 0, stream>>>((unsigned*)stat, 513);
    cvt_input<<<dim3(8192), 256, 0, stream>>>(x, Xp, 32, 256, 256);

    auto prep = [&](const float* w, _Float16* o, int O, int I, int KH, int KW, int CP) {
        const int total = O * KH * KW * CP;
        prep_conv_w<<<dim3((total + 255) / 256), 256, 0, stream>>>(w, o, O, I, KH, KW, CP);
    };
    prep(we1,   W1,   64, 3,   4, 4, 4);
    prep(we2,   W2,   128, 64, 4, 4, 64);
    prep(we3,   W3,   128, 128, 3, 3, 128);
    prep(er0w3, ER0A, 32, 128, 3, 3, 128);
    prep(er0w1, ER0B, 128, 32, 1, 1, 32);
    prep(er1w3, ER1A, 32, 128, 3, 3, 128);
    prep(er1w1, ER1B, 128, 32, 1, 1, 32);
    prep(wpq,   WPQ,  64, 128, 1, 1, 128);
    prep(wd1,   WD1,  128, 64, 3, 3, 64);
    prep(dr0w3, DR0A, 32, 128, 3, 3, 128);
    prep(dr0w1, DR0B, 128, 32, 1, 1, 32);
    prep(dr1w3, DR1A, 32, 128, 3, 3, 128);
    prep(dr1w1, DR1B, 128, 32, 1, 1, 32);
    prep_convt_w<<<dim3((4 * 64 * 4 * 128 + 255) / 256), 256, 0, stream>>>(wt1, WT1, 128, 64, 64);
    prep_convt_w<<<dim3((4 * 16 * 4 * 64 + 255) / 256), 256, 0, stream>>>(wt2, WT2, 64, 3, 16);
    prep_codebook<<<dim3(2), 256, 0, stream>>>(cbf, CBH, cnorm);

    // --- encoder ---
    conv_wmma<4, 4, 4, 64, 64, 128, 2, 1, false, true, false>
        <<<dim3(4096, 1), 256, 0, stream>>>(Xp, W1, be1, nullptr, A1, 32, 256, 256, 128, 128);
    conv_wmma<4, 4, 64, 128, 64, 128, 2, 1, false, true, false>
        <<<dim3(1024, 2), 256, 0, stream>>>(A1, W2, be2, nullptr, A2, 32, 128, 128, 64, 64);
    conv_wmma<3, 3, 128, 128, 64, 128, 1, 1, false, false, false>
        <<<dim3(1024, 2), 256, 0, stream>>>(A2, W3, be3, nullptr, E, 32, 64, 64, 64, 64);
    // enc residual stack
    conv_wmma<3, 3, 128, 32, 32, 128, 1, 1, true, false, false>
        <<<dim3(1024, 1), 256, 0, stream>>>(E, ER0A, nullptr, nullptr, Tt, 32, 64, 64, 64, 64);
    conv_wmma<1, 1, 32, 128, 64, 128, 1, 0, true, false, true>
        <<<dim3(1024, 2), 256, 0, stream>>>(Tt, ER0B, nullptr, E, E, 32, 64, 64, 64, 64);
    conv_wmma<3, 3, 128, 32, 32, 128, 1, 1, true, false, false>
        <<<dim3(1024, 1), 256, 0, stream>>>(E, ER1A, nullptr, nullptr, Tt, 32, 64, 64, 64, 64);
    conv_wmma<1, 1, 32, 128, 64, 128, 1, 0, true, false, true>
        <<<dim3(1024, 2), 256, 0, stream>>>(Tt, ER1B, nullptr, E, E, 32, 64, 64, 64, 64);
    // pre-VQ 1x1 (input is relu(trunk))
    conv_wmma<1, 1, 128, 64, 64, 128, 1, 0, true, false, false>
        <<<dim3(1024, 1), 256, 0, stream>>>(E, WPQ, bpq, nullptr, Z, 32, 64, 64, 64, 64);

    // --- VQ ---
    vq_kernel<<<dim3(8192), 256, 0, stream>>>(Z, CBH, cnorm, Q, idxf, counts, sse);
    vq_finalize<<<dim3(1), 512, 0, stream>>>(counts, sse, out, perp,
                                             1.f / 131072.f, 1.f / 8388608.f);

    // --- decoder ---
    conv_wmma<3, 3, 64, 128, 64, 128, 1, 1, false, false, false>
        <<<dim3(1024, 2), 256, 0, stream>>>(Q, WD1, bd1, nullptr, D0, 32, 64, 64, 64, 64);
    conv_wmma<3, 3, 128, 32, 32, 128, 1, 1, true, false, false>
        <<<dim3(1024, 1), 256, 0, stream>>>(D0, DR0A, nullptr, nullptr, Tt, 32, 64, 64, 64, 64);
    conv_wmma<1, 1, 32, 128, 64, 128, 1, 0, true, false, true>
        <<<dim3(1024, 2), 256, 0, stream>>>(Tt, DR0B, nullptr, D0, D0, 32, 64, 64, 64, 64);
    conv_wmma<3, 3, 128, 32, 32, 128, 1, 1, true, false, false>
        <<<dim3(1024, 1), 256, 0, stream>>>(D0, DR1A, nullptr, nullptr, Tt, 32, 64, 64, 64, 64);
    conv_wmma<1, 1, 32, 128, 64, 128, 1, 0, true, false, true>
        <<<dim3(1024, 2), 256, 0, stream>>>(Tt, DR1B, nullptr, D0, D0, 32, 64, 64, 64, 64);
    // transposed convs (parity-decomposed)
    convt_wmma<128, 64, 64, 64, true, true, false>
        <<<dim3(4096, 2, 1), 256, 0, stream>>>(D0, WT1, bt1, T1, nullptr, 32, 64, 64, 128, 128);
    convt_wmma<64, 3, 16, 128, false, false, true>
        <<<dim3(8192, 2, 1), 256, 0, stream>>>(T1, WT2, bt2, nullptr, recon, 32, 128, 128, 256, 256);
}
